// ODERNNEncoder_19078244729127
// MI455X (gfx1250) — matrix operand order repaired and checked by reading, not compiled, not run on hardware
//
#include <hip/hip_runtime.h>
#include <hip/hip_bf16.h>
#include <stdint.h>

// ODE-RNN encoder for MI455X (gfx1250).
// Single persistent kernel: 64 blocks x 16 batch-rows, all 64 timesteps on-chip.
// All matmuls via v_wmma_f32_16x16x32_f16; activations stored f16 row-major so
// A-fragments are two unconditional b128 loads; weights pre-packed in B-fragment
// order (two b128 loads). h and kacc of the RK4 live in registers.
// K-loop at unroll 2: 2-deep pipeline overlaps next iteration's fragment loads
// with current WMMAs (unroll 1 serializes on load latency; full unroll spills).

typedef __attribute__((ext_vector_type(16))) _Float16 v16h;
typedef __attribute__((ext_vector_type(8)))  float    v8f;

#define BATCH  1024
#define HID    512
#define OBS    8
#define TSTEPS 64
#define LN_EPS 1e-5f
#define ROWS_PER_BLOCK 16

__device__ __forceinline__ uint16_t f2h(float v) {
    union { _Float16 h; uint16_t u; } c; c.h = (_Float16)v; return c.u;
}
__device__ __forceinline__ uint32_t pack2h(float a, float b) {
    union { _Float16 h[2]; uint32_t u; } p;
    p.h[0] = (_Float16)a; p.h[1] = (_Float16)b;
    return p.u;
}

// ---------------------------------------------------------------------------
// Weight pre-pack: f32 [K,N] row-major -> f16 in WMMA B-fragment order.
// dword index = ((ntile*KB + kblock)*32 + lane)*8 + g
//   lane = 16*khalf + ncol ; dword g holds K = kb*32 + khalf*16 + 2g, 2g+1
// ---------------------------------------------------------------------------
__global__ void pack_weights(const float* __restrict__ src, uint32_t* __restrict__ dst,
                             int K, int N, int KB) {
    int idx = blockIdx.x * blockDim.x + threadIdx.x;
    int total = (N / 16) * KB * 256;
    if (idx >= total) return;
    int nt   = idx / (KB * 256);
    int rem  = idx % (KB * 256);
    int kb   = rem / 256;
    int r2   = rem % 256;
    int lane = r2 / 8;
    int g    = r2 % 8;
    int kh   = lane / 16;
    int ncol = nt * 16 + (lane % 16);
    int k0   = kb * 32 + kh * 16 + 2 * g;
    float v0 = (k0     < K) ? src[(size_t)k0       * N + ncol] : 0.f;
    float v1 = (k0 + 1 < K) ? src[(size_t)(k0 + 1) * N + ncol] : 0.f;
    dst[idx] = pack2h(v0, v1);
}

// xs f32 [T*B, 8] -> f16 [T*B, 32], zero-padded K
__global__ void cvt_xs(const float* __restrict__ xs, uint16_t* __restrict__ xs16) {
    int i = blockIdx.x * blockDim.x + threadIdx.x;
    if (i >= TSTEPS * BATCH * 32) return;
    int c = i & 31, row = i >> 5;
    float v = (c < OBS) ? xs[(size_t)row * OBS + c] : 0.f;
    xs16[i] = f2h(v);
}

// ---------------------------------------------------------------------------
// Core WMMA tile loop: wave computes 16 rows x NT*16 cols, K = KB*32.
// A16: f16 row-major (block's row 0), lda in elements. Wp: packed B fragments.
// ---------------------------------------------------------------------------
template<int NT>
__device__ __forceinline__ void mma_loop(const uint16_t* __restrict__ A16, int lda,
                                         const uint32_t* __restrict__ Wp, int KB, int nt0,
                                         int lane, v8f (&acc)[NT]) {
    const int lo = lane & 15, hi = lane >> 4;
    const uint16_t* arow = A16 + (size_t)lo * lda + 16 * hi;
    #pragma unroll 2
    for (int kb = 0; kb < KB; ++kb) {
        union FragU { uint32_t u[8]; v16h h; } afr, bfr[NT];
        const uint4* ap = reinterpret_cast<const uint4*>(arow + kb * 32);
        uint4 a0 = ap[0];           // K = kb*32 + 16*hi + 0..7
        uint4 a1 = ap[1];           // K = kb*32 + 16*hi + 8..15
        afr.u[0] = a0.x; afr.u[1] = a0.y; afr.u[2] = a0.z; afr.u[3] = a0.w;
        afr.u[4] = a1.x; afr.u[5] = a1.y; afr.u[6] = a1.z; afr.u[7] = a1.w;
        #pragma unroll
        for (int nt = 0; nt < NT; ++nt) {
            const uint4* p = reinterpret_cast<const uint4*>(
                Wp + (((size_t)(nt0 + nt) * KB + kb) * 32 + lane) * 8);
            uint4 q0 = p[0], q1 = p[1];
            bfr[nt].u[0] = q0.x; bfr[nt].u[1] = q0.y; bfr[nt].u[2] = q0.z; bfr[nt].u[3] = q0.w;
            bfr[nt].u[4] = q1.x; bfr[nt].u[5] = q1.y; bfr[nt].u[6] = q1.z; bfr[nt].u[7] = q1.w;
        }
        if (kb + 1 < KB)
            __builtin_prefetch(Wp + (((size_t)nt0 * KB + kb + 1) * 32 + lane) * 8, 0, 0);
        #pragma unroll
        for (int nt = 0; nt < NT; ++nt)
            acc[nt] = __builtin_amdgcn_wmma_f32_16x16x32_f16(
                false, afr.h, false, bfr[nt].h, (short)0, acc[nt], false, false);
    }
}

__device__ __forceinline__ void acc_zero(v8f (&acc)[4]) {
    #pragma unroll
    for (int nt = 0; nt < 4; ++nt)
        #pragma unroll
        for (int e = 0; e < 8; ++e) acc[nt][e] = 0.f;
}

__device__ __forceinline__ void bias_add(v8f (&acc)[4], int w, int lane,
                                         const float* __restrict__ bias,
                                         const float* __restrict__ extra, float tv) {
    int lo = lane & 15;
    #pragma unroll
    for (int nt = 0; nt < 4; ++nt) {
        int col = w * 64 + nt * 16 + lo;
        float be = bias[col];
        if (extra) be += tv * extra[col];
        #pragma unroll
        for (int r = 0; r < 8; ++r) acc[nt][r] += be;
    }
}

// LayerNorm (over 512 cols) + LeakyReLU(0.01), in place on acc.
__device__ __forceinline__ void ln_leaky(v8f (&acc)[4], int w, int lane, int tid,
                                         const float* __restrict__ g, const float* __restrict__ b,
                                         float (*red_s)[16], float (*red_q)[16],
                                         float* s_mean, float* s_rstd) {
    int lo = lane & 15, hi = lane >> 4;
    #pragma unroll
    for (int r = 0; r < 8; ++r) {
        float s = 0.f, q = 0.f;
        #pragma unroll
        for (int nt = 0; nt < 4; ++nt) { float v = acc[nt][r]; s += v; q += v * v; }
        #pragma unroll
        for (int off = 1; off < 16; off <<= 1) {
            s += __shfl_xor(s, off, 32);
            q += __shfl_xor(q, off, 32);
        }
        if (lo == 0) { red_s[w][8 * hi + r] = s; red_q[w][8 * hi + r] = q; }
    }
    __syncthreads();
    if (tid < 16) {
        float s = 0.f, q = 0.f;
        #pragma unroll
        for (int ww = 0; ww < 8; ++ww) { s += red_s[ww][tid]; q += red_q[ww][tid]; }
        float mean = s * (1.f / 512.f);
        float var  = q * (1.f / 512.f) - mean * mean;
        s_mean[tid] = mean;
        s_rstd[tid] = rsqrtf(var + LN_EPS);
    }
    __syncthreads();
    #pragma unroll
    for (int r = 0; r < 8; ++r) {
        int row = 8 * hi + r;
        float mean = s_mean[row], rstd = s_rstd[row];
        #pragma unroll
        for (int nt = 0; nt < 4; ++nt) {
            int col = w * 64 + nt * 16 + lo;
            float x = (acc[nt][r] - mean) * rstd * g[col] + b[col];
            acc[nt][r] = x > 0.f ? x : 0.01f * x;
        }
    }
}

// store wave's tile into f16 row-major [16][512] block buffer
__device__ __forceinline__ void store_tile16(uint16_t* __restrict__ out,
                                             const v8f (&acc)[4], int w, int lane) {
    int lo = lane & 15, hi = lane >> 4;
    #pragma unroll
    for (int nt = 0; nt < 4; ++nt) {
        int col = w * 64 + nt * 16 + lo;
        #pragma unroll
        for (int r = 0; r < 8; ++r)
            out[(size_t)(8 * hi + r) * HID + col] = f2h(acc[nt][r]);
    }
}

__device__ __forceinline__ void phase_sync() {
    __threadfence_block();
    __syncthreads();
}

// ---------------------------------------------------------------------------
__global__ __launch_bounds__(256, 1)
void odernn_persistent(
    const uint16_t* __restrict__ xs16,
    const uint32_t* __restrict__ wp_obs, const uint32_t* __restrict__ wp_W1,
    const uint32_t* __restrict__ wp_W2,  const uint32_t* __restrict__ wp_Wout,
    const uint32_t* __restrict__ wp_ih,  const uint32_t* __restrict__ wp_hh,
    const float* __restrict__ obs_b,  const float* __restrict__ obs_g,  const float* __restrict__ obs_bb,
    const float* __restrict__ b1, const float* __restrict__ w1row,
    const float* __restrict__ g1, const float* __restrict__ bb1,
    const float* __restrict__ b2, const float* __restrict__ g2, const float* __restrict__ bb2,
    const float* __restrict__ bout,
    const float* __restrict__ b_ih, const float* __restrict__ b_hh,
    uint16_t* __restrict__ y16g, uint16_t* __restrict__ z1g, uint16_t* __restrict__ z2g,
    uint16_t* __restrict__ h16g, uint16_t* __restrict__ hp16g,
    float* __restrict__ hprevg, float* __restrict__ gig, float* __restrict__ ghg,
    float* __restrict__ dout)
{
    __shared__ float red_s[8][16];
    __shared__ float red_q[8][16];
    __shared__ float s_mean[16];
    __shared__ float s_rstd[16];

    const int tid  = threadIdx.x;
    const int w    = tid >> 5;
    const int lane = tid & 31;
    const int lo   = lane & 15;
    const int hi   = lane >> 4;
    const int rb   = blockIdx.x;              // row block: rows [rb*16, rb*16+16)

    // block-local regions
    uint16_t* y16  = y16g  + (size_t)rb * ROWS_PER_BLOCK * HID;
    uint16_t* z1   = z1g   + (size_t)rb * ROWS_PER_BLOCK * HID;
    uint16_t* z2   = z2g   + (size_t)rb * ROWS_PER_BLOCK * HID;
    uint16_t* h16  = h16g  + (size_t)rb * ROWS_PER_BLOCK * HID;
    uint16_t* hp16 = hp16g + (size_t)rb * ROWS_PER_BLOCK * HID;
    float*    hp   = hprevg + (size_t)rb * ROWS_PER_BLOCK * HID;
    float*    gi   = gig   + (size_t)rb * ROWS_PER_BLOCK * 3 * HID;
    float*    gh   = ghg   + (size_t)rb * ROWS_PER_BLOCK * 3 * HID;

    // zero initial hidden state
    for (int i = tid; i < ROWS_PER_BLOCK * HID; i += 256) { hp[i] = 0.f; hp16[i] = 0; }
    phase_sync();

    float hreg[4][8];   // ODE state h, element (row = 8*hi+r, col = w*64+nt*16+lo)
    float kreg[4][8];   // RK4 accumulator k1 + 2k2 + 2k3
    v8f acc[4];

    #pragma unroll 1
    for (int t = 0; t < TSTEPS; ++t) {
        // ---- obs -> hidden: LN + leaky ----
        acc_zero(acc);
        mma_loop<4>(xs16 + ((size_t)t * BATCH + (size_t)rb * ROWS_PER_BLOCK) * 32, 32,
                    wp_obs, 1, w * 4, lane, acc);
        bias_add(acc, w, lane, obs_b, nullptr, 0.f);
        ln_leaky(acc, w, lane, tid, obs_g, obs_bb, red_s, red_q, s_mean, s_rstd);
        #pragma unroll
        for (int nt = 0; nt < 4; ++nt)
            #pragma unroll
            for (int r = 0; r < 8; ++r) hreg[nt][r] = acc[nt][r];
        store_tile16(h16, acc, w, lane);
        phase_sync();

        // ---- RK4 over t in [0,1], 4 steps ----
        #pragma unroll 1
        for (int s = 0; s < 4; ++s) {
            float t0 = (float)s * 0.25f;
            #pragma unroll 1
            for (int st = 0; st < 4; ++st) {
                const uint16_t* cur = (st == 0) ? h16 : y16;
                float tv = t0 + ((st == 0) ? 0.f : (st == 3) ? 0.25f : 0.125f);
                // layer 1 (time column folded into bias)
                acc_zero(acc);
                mma_loop<4>(cur, HID, wp_W1, 16, w * 4, lane, acc);
                bias_add(acc, w, lane, b1, w1row, tv);
                ln_leaky(acc, w, lane, tid, g1, bb1, red_s, red_q, s_mean, s_rstd);
                store_tile16(z1, acc, w, lane);
                phase_sync();
                // layer 2
                acc_zero(acc);
                mma_loop<4>(z1, HID, wp_W2, 16, w * 4, lane, acc);
                bias_add(acc, w, lane, b2, nullptr, 0.f);
                ln_leaky(acc, w, lane, tid, g2, bb2, red_s, red_q, s_mean, s_rstd);
                store_tile16(z2, acc, w, lane);
                phase_sync();
                // layer 3 (linear out) + RK4 stage update in registers
                acc_zero(acc);
                mma_loop<4>(z2, HID, wp_Wout, 16, w * 4, lane, acc);
                bias_add(acc, w, lane, bout, nullptr, 0.f);
                if (st == 0) {
                    #pragma unroll
                    for (int nt = 0; nt < 4; ++nt)
                        #pragma unroll
                        for (int r = 0; r < 8; ++r) {
                            float kv = acc[nt][r];
                            kreg[nt][r] = kv;
                            int col = w * 64 + nt * 16 + lo;
                            y16[(size_t)(8 * hi + r) * HID + col] = f2h(hreg[nt][r] + 0.125f * kv);
                        }
                } else if (st < 3) {
                    float cy = (st == 2) ? 0.25f : 0.125f;
                    #pragma unroll
                    for (int nt = 0; nt < 4; ++nt)
                        #pragma unroll
                        for (int r = 0; r < 8; ++r) {
                            float kv = acc[nt][r];
                            kreg[nt][r] += 2.f * kv;
                            int col = w * 64 + nt * 16 + lo;
                            y16[(size_t)(8 * hi + r) * HID + col] = f2h(hreg[nt][r] + cy * kv);
                        }
                } else {
                    #pragma unroll
                    for (int nt = 0; nt < 4; ++nt)
                        #pragma unroll
                        for (int r = 0; r < 8; ++r) {
                            hreg[nt][r] += (0.25f / 6.f) * (kreg[nt][r] + acc[nt][r]);
                            int col = w * 64 + nt * 16 + lo;
                            h16[(size_t)(8 * hi + r) * HID + col] = f2h(hreg[nt][r]);
                        }
                }
                phase_sync();
            }
        }

        // ---- GRU: gi = h_ode@W_ih + b_ih ; gh = h_prev@W_hh + b_hh ----
        #pragma unroll 1
        for (int c = 0; c < 3; ++c) {
            acc_zero(acc);
            mma_loop<4>(h16, HID, wp_ih, 16, c * 32 + w * 4, lane, acc);
            #pragma unroll
            for (int nt = 0; nt < 4; ++nt) {
                int col = c * 512 + w * 64 + nt * 16 + lo;
                float be = b_ih[col];
                #pragma unroll
                for (int r = 0; r < 8; ++r)
                    gi[(size_t)(8 * hi + r) * (3 * HID) + col] = acc[nt][r] + be;
            }
        }
        #pragma unroll 1
        for (int c = 0; c < 3; ++c) {
            acc_zero(acc);
            mma_loop<4>(hp16, HID, wp_hh, 16, c * 32 + w * 4, lane, acc);
            #pragma unroll
            for (int nt = 0; nt < 4; ++nt) {
                int col = c * 512 + w * 64 + nt * 16 + lo;
                float be = b_hh[col];
                #pragma unroll
                for (int r = 0; r < 8; ++r)
                    gh[(size_t)(8 * hi + r) * (3 * HID) + col] = acc[nt][r] + be;
            }
        }
        phase_sync();

        // ---- GRU gates (element-wise, rows local to block) ----
        #pragma unroll 1
        for (int i = tid; i < ROWS_PER_BLOCK * HID; i += 256) {
            int r = i >> 9, j = i & 511;
            size_t gb = (size_t)r * (3 * HID);
            float ir = gi[gb + j], iz = gi[gb + HID + j], in_ = gi[gb + 2 * HID + j];
            float hr = gh[gb + j], hz = gh[gb + HID + j], hn  = gh[gb + 2 * HID + j];
            float rg = 1.f / (1.f + expf(-(ir + hr)));
            float zg = 1.f / (1.f + expf(-(iz + hz)));
            float ng = tanhf(in_ + rg * hn);
            float hn2 = (1.f - zg) * ng + zg * hp[i];
            hp[i] = hn2;
            hp16[i] = f2h(hn2);
        }
        phase_sync();
    }

    // final hidden -> output
    for (int i = tid; i < ROWS_PER_BLOCK * HID; i += 256)
        dout[(size_t)rb * ROWS_PER_BLOCK * HID + i] = hp[i];
}

// ---------------------------------------------------------------------------
extern "C" void kernel_launch(void* const* d_in, const int* in_sizes, int n_in,
                              void* d_out, int out_size, void* d_ws, size_t ws_size,
                              hipStream_t stream) {
    const float* xs       = (const float*)d_in[0];
    const float* obs_W    = (const float*)d_in[1];
    const float* obs_b    = (const float*)d_in[2];
    const float* obs_ln_g = (const float*)d_in[3];
    const float* obs_ln_b = (const float*)d_in[4];
    const float* ode_W1   = (const float*)d_in[5];
    const float* ode_b1   = (const float*)d_in[6];
    const float* ode_ln1g = (const float*)d_in[7];
    const float* ode_ln1b = (const float*)d_in[8];
    const float* ode_W2   = (const float*)d_in[9];
    const float* ode_b2   = (const float*)d_in[10];
    const float* ode_ln2g = (const float*)d_in[11];
    const float* ode_ln2b = (const float*)d_in[12];
    const float* ode_Wout = (const float*)d_in[13];
    const float* ode_bout = (const float*)d_in[14];
    const float* W_ih     = (const float*)d_in[15];
    const float* b_ih     = (const float*)d_in[16];
    const float* W_hh     = (const float*)d_in[17];
    const float* b_hh     = (const float*)d_in[18];
    (void)in_sizes; (void)n_in; (void)out_size; (void)ws_size;

    char* ws = (char*)d_ws;
    auto alloc = [&](size_t bytes) -> char* {
        char* p = ws; ws += (bytes + 255) & ~(size_t)255; return p;
    };
    uint32_t* wp_obs  = (uint32_t*)alloc((size_t)32 * 1  * 256 * 4);
    uint32_t* wp_W1   = (uint32_t*)alloc((size_t)32 * 16 * 256 * 4);
    uint32_t* wp_W2   = (uint32_t*)alloc((size_t)32 * 16 * 256 * 4);
    uint32_t* wp_Wout = (uint32_t*)alloc((size_t)32 * 16 * 256 * 4);
    uint32_t* wp_ih   = (uint32_t*)alloc((size_t)96 * 16 * 256 * 4);
    uint32_t* wp_hh   = (uint32_t*)alloc((size_t)96 * 16 * 256 * 4);
    uint16_t* xs16    = (uint16_t*)alloc((size_t)TSTEPS * BATCH * 32 * 2);
    uint16_t* y16g    = (uint16_t*)alloc((size_t)BATCH * HID * 2);
    uint16_t* z1g     = (uint16_t*)alloc((size_t)BATCH * HID * 2);
    uint16_t* z2g     = (uint16_t*)alloc((size_t)BATCH * HID * 2);
    uint16_t* h16g    = (uint16_t*)alloc((size_t)BATCH * HID * 2);
    uint16_t* hp16g   = (uint16_t*)alloc((size_t)BATCH * HID * 2);
    float*    hprevg  = (float*)alloc((size_t)BATCH * HID * 4);
    float*    gig     = (float*)alloc((size_t)BATCH * 3 * HID * 4);
    float*    ghg     = (float*)alloc((size_t)BATCH * 3 * HID * 4);

    auto packW = [&](const float* src, uint32_t* dst, int K, int N) {
        int KB = (K + 31) / 32;
        int total = (N / 16) * KB * 256;
        pack_weights<<<dim3((total + 255) / 256), dim3(256), 0, stream>>>(src, dst, K, N, KB);
    };
    packW(obs_W,    wp_obs,  OBS, HID);
    packW(ode_W1,   wp_W1,   HID, HID);   // row 512 (time) folded into bias
    packW(ode_W2,   wp_W2,   HID, HID);
    packW(ode_Wout, wp_Wout, HID, HID);
    packW(W_ih,     wp_ih,   HID, 3 * HID);
    packW(W_hh,     wp_hh,   HID, 3 * HID);

    int nxs = TSTEPS * BATCH * 32;
    cvt_xs<<<dim3((nxs + 255) / 256), dim3(256), 0, stream>>>(xs, xs16);

    const float* w1row = ode_W1 + (size_t)512 * 512;
    odernn_persistent<<<dim3(BATCH / ROWS_PER_BLOCK), dim3(256), 0, stream>>>(
        xs16, wp_obs, wp_W1, wp_W2, wp_Wout, wp_ih, wp_hh,
        obs_b, obs_ln_g, obs_ln_b,
        ode_b1, w1row, ode_ln1g, ode_ln1b,
        ode_b2, ode_ln2g, ode_ln2b,
        ode_bout, b_ih, b_hh,
        y16g, z1g, z2g, h16g, hp16g, hprevg, gig, ghg,
        (float*)d_out);
}
